// BertSelfAttention_80126909874551
// MI455X (gfx1250) — matrix operand order repaired and checked by reading, not compiled, not run on hardware
//
#include <hip/hip_runtime.h>

// ---------------------------------------------------------------------------
// BERT self-attention for MI455X (gfx1250), wave32 + WMMA bf16.
//   B=2, S=2048, HIDDEN=1024, 16 heads x 64 dim.
// Kernel 1: fused QKV projection GEMMs. 128x64 output tile per block,
//           double-buffered LDS staging (one barrier per K-chunk), each wave
//           owns a 2x2 grid of 16x16 fp32 accumulators (4 WMMA / chunk).
//           Writes Q (pre-scaled by 1/sqrt(64)*log2 e) / K as [BH,S,D] and
//           V transposed [BH,D,S] bf16 into workspace.
// Kernel 2: flash attention, one wave per 16-row q-tile, 64-wide K/V tiles
//           (16 WMMA per online-softmax pass), half-wave shfl reductions,
//           global_prefetch on the next K/V tiles.
// ---------------------------------------------------------------------------

typedef __attribute__((ext_vector_type(16))) __bf16        v16bf;
typedef __attribute__((ext_vector_type(8)))  float         v8f;
typedef __attribute__((ext_vector_type(8)))  unsigned int  v8u;
typedef __attribute__((ext_vector_type(4)))  unsigned int  v4u;

union AB {            // one WMMA 16-bit A/B fragment: 16 bf16 = 8 dwords
  v16bf bf;
  v8u   u8;
  v4u   u4[2];
};

#define HID   1024
#define NH    16
#define HD    64
#define BATCH 2
#define SEQ   2048
#define TOK   (BATCH * SEQ)

#if defined(__AMDGCN__) && __has_builtin(__builtin_amdgcn_exp2f)
#define EXP2F __builtin_amdgcn_exp2f
#else
#define EXP2F exp2f
#endif

__device__ inline v8f wmma_bf16(v16bf a, v16bf b, v8f c) {
  // (neg_a, A, neg_b, B, c_mod, C, reuse_a, reuse_b)
  return __builtin_amdgcn_wmma_f32_16x16x32_bf16(false, a, false, b,
                                                 (short)0, c, false, false);
}

// max / sum over the 16-lane half-group (masks < 16 stay inside the half).
__device__ inline float hg_max(float v) {
  v = fmaxf(v, __shfl_xor(v, 1, 32));
  v = fmaxf(v, __shfl_xor(v, 2, 32));
  v = fmaxf(v, __shfl_xor(v, 4, 32));
  v = fmaxf(v, __shfl_xor(v, 8, 32));
  return v;
}
__device__ inline float hg_sum(float v) {
  v += __shfl_xor(v, 1, 32);
  v += __shfl_xor(v, 2, 32);
  v += __shfl_xor(v, 4, 32);
  v += __shfl_xor(v, 8, 32);
  return v;
}

// ---------------------------------------------------------------------------
// Kernel 1: Y = X*W + b for W in {Wq,Wk,Wv}.
// Block = 8 waves -> 128x64 tile; wave w owns rows {2(w>>1),2(w>>1)+1} x
// cols {2(w&1),2(w&1)+1} sub-tiles. LDS ping-pong staging, 1 barrier/chunk.
// ---------------------------------------------------------------------------
__global__ __launch_bounds__(256, 2)
void qkv_proj_kernel(const float* __restrict__ X,
                     const float* __restrict__ Wq, const float* __restrict__ bq,
                     const float* __restrict__ Wk, const float* __restrict__ bk,
                     const float* __restrict__ Wv, const float* __restrict__ bv,
                     __bf16* __restrict__ Qb, __bf16* __restrict__ Kb,
                     __bf16* __restrict__ Vt) {
  __shared__ __attribute__((aligned(32))) __bf16 Xs[2][128][40]; // 128x32 tile
  __shared__ __attribute__((aligned(32))) __bf16 Wt[2][64][48];  // 32x64 ^T

  const int tid   = threadIdx.x;
  const int lane  = tid & 31;
  const int wv    = tid >> 5;
  const int group = lane >> 4;
  const int ln    = lane & 15;

  const int col0 = blockIdx.x * 64;
  const int t0   = blockIdx.y * 128;
  const int proj = blockIdx.z;

  const float* W    = (proj == 0) ? Wq : (proj == 1) ? Wk : Wv;
  const float* bias = (proj == 0) ? bq : (proj == 1) ? bk : bv;

  const int rs = (wv >> 1) * 2;    // first of 2 row sub-tiles (of 8)
  const int cs = (wv & 1) * 2;     // first of 2 col sub-tiles (of 4)

  // staging assignment: X tile 128x32 (16 f32/thread), W tile 32x64 (8/thread)
  const int xr = tid >> 1, xc = (tid & 1) * 16;
  const int wr = tid >> 3, wc = (tid & 7) * 8;
  const float* Xsrc = X + (size_t)(t0 + xr) * HID + xc;
  const float* Wsrc = W + (size_t)wr * HID + col0 + wc;

  float4 xf0, xf1, xf2, xf3, wf0, wf1;
  auto load_stage = [&](int kc) {
    const float* xs = Xsrc + kc;
    xf0 = *(const float4*)(xs);      xf1 = *(const float4*)(xs + 4);
    xf2 = *(const float4*)(xs + 8);  xf3 = *(const float4*)(xs + 12);
    const float* ws = Wsrc + (size_t)kc * HID;
    wf0 = *(const float4*)(ws);      wf1 = *(const float4*)(ws + 4);
  };
  auto store_stage = [&](int bsel) {
    __bf16* xd = &Xs[bsel][xr][xc];
    xd[0]  = (__bf16)xf0.x; xd[1]  = (__bf16)xf0.y;
    xd[2]  = (__bf16)xf0.z; xd[3]  = (__bf16)xf0.w;
    xd[4]  = (__bf16)xf1.x; xd[5]  = (__bf16)xf1.y;
    xd[6]  = (__bf16)xf1.z; xd[7]  = (__bf16)xf1.w;
    xd[8]  = (__bf16)xf2.x; xd[9]  = (__bf16)xf2.y;
    xd[10] = (__bf16)xf2.z; xd[11] = (__bf16)xf2.w;
    xd[12] = (__bf16)xf3.x; xd[13] = (__bf16)xf3.y;
    xd[14] = (__bf16)xf3.z; xd[15] = (__bf16)xf3.w;
    Wt[bsel][wc + 0][wr] = (__bf16)wf0.x;
    Wt[bsel][wc + 1][wr] = (__bf16)wf0.y;
    Wt[bsel][wc + 2][wr] = (__bf16)wf0.z;
    Wt[bsel][wc + 3][wr] = (__bf16)wf0.w;
    Wt[bsel][wc + 4][wr] = (__bf16)wf1.x;
    Wt[bsel][wc + 5][wr] = (__bf16)wf1.y;
    Wt[bsel][wc + 6][wr] = (__bf16)wf1.z;
    Wt[bsel][wc + 7][wr] = (__bf16)wf1.w;
  };

  const v8f vzero = {0.f, 0.f, 0.f, 0.f, 0.f, 0.f, 0.f, 0.f};
  v8f a00 = vzero, a01 = vzero, a10 = vzero, a11 = vzero;

  auto compute = [&](int bsel) {
    AB aF0, aF1, bF0, bF1;
    aF0.u4[0] = *(const v4u*)&Xs[bsel][(rs + 0) * 16 + ln][group * 8];
    aF0.u4[1] = *(const v4u*)&Xs[bsel][(rs + 0) * 16 + ln][16 + group * 8];
    aF1.u4[0] = *(const v4u*)&Xs[bsel][(rs + 1) * 16 + ln][group * 8];
    aF1.u4[1] = *(const v4u*)&Xs[bsel][(rs + 1) * 16 + ln][16 + group * 8];
    bF0.u4[0] = *(const v4u*)&Wt[bsel][(cs + 0) * 16 + ln][group * 8];
    bF0.u4[1] = *(const v4u*)&Wt[bsel][(cs + 0) * 16 + ln][16 + group * 8];
    bF1.u4[0] = *(const v4u*)&Wt[bsel][(cs + 1) * 16 + ln][group * 8];
    bF1.u4[1] = *(const v4u*)&Wt[bsel][(cs + 1) * 16 + ln][16 + group * 8];
    a00 = wmma_bf16(aF0.bf, bF0.bf, a00);
    a01 = wmma_bf16(aF0.bf, bF1.bf, a01);
    a10 = wmma_bf16(aF1.bf, bF0.bf, a10);
    a11 = wmma_bf16(aF1.bf, bF1.bf, a11);
  };

  // software pipeline: chunk c lives in LDS buffer (c & 1)
  load_stage(0);
  store_stage(0);
  __syncthreads();
  for (int it = 1; it < HID / 32; ++it) {
    load_stage(it * 32);        // global prefetch of chunk `it`
    compute((it - 1) & 1);      // WMMA on chunk `it-1`
    store_stage(it & 1);        // commit chunk `it` to the other buffer
    __syncthreads();
  }
  compute((HID / 32 - 1) & 1);

  // fold softmax scale and log2(e) into Q so attention uses v_exp_f32 (exp2)
  const float qscale = 0.125f * 1.44269504088896340736f;
  const v8f accs[2][2] = {{a00, a01}, {a10, a11}};
  for (int rr = 0; rr < 2; ++rr)
    for (int cc = 0; cc < 2; ++cc) {
      const v8f acc = accs[rr][cc];
      const int col = col0 + (cs + cc) * 16 + ln;
      const int h = col >> 6, d = col & 63;
      const float bb = bias[col];
      for (int i = 0; i < 8; ++i) {
        const int tok = t0 + (rs + rr) * 16 + group * 8 + i;
        const int b = tok >> 11, s = tok & (SEQ - 1);
        const int bh = b * NH + h;
        const float v = acc[i] + bb;
        if (proj == 0)
          Qb[((size_t)(bh * SEQ + s) << 6) + d] = (__bf16)(v * qscale);
        else if (proj == 1)
          Kb[((size_t)(bh * SEQ + s) << 6) + d] = (__bf16)v;
        else
          Vt[(size_t)(bh * HD + d) * SEQ + s] = (__bf16)v;  // V transposed
      }
    }
}

// ---------------------------------------------------------------------------
// Kernel 2: flash attention. One wave per (b,h, 16-row q tile); 64-col K/V
// tiles; 16 WMMAs per tile (8 for Q*K^T over D=64, 8 for P*V over D=64).
// ---------------------------------------------------------------------------
__global__ __launch_bounds__(256, 2)
void attn_kernel(const __bf16* __restrict__ Qb, const __bf16* __restrict__ Kb,
                 const __bf16* __restrict__ Vt, float* __restrict__ Out) {
  __shared__ __attribute__((aligned(32))) __bf16 Ptile[8][16][72]; // per wave

  const int tid   = threadIdx.x;
  const int lane  = tid & 31;
  const int wv    = tid >> 5;
  const int group = lane >> 4;
  const int ln    = lane & 15;

  const int gw = blockIdx.x * 8 + wv;   // global wave id: 32 bh * 128 qtiles
  const int bh = gw >> 7;
  const int q0 = (gw & 127) << 4;

  // ---- Q fragments (registers for entire kernel), A layout pair loads ----
  const __bf16* Qrow = Qb + (size_t)(bh * SEQ + q0 + ln) * HD;
  AB aq[2];
  for (int c = 0; c < 2; ++c)
    for (int p = 0; p < 8; ++p) {
      const int d = c * 32 + group * 16 + (p << 1);
      aq[c].u8[p] = *(const unsigned int*)(Qrow + d);
    }

  const __bf16* Kbase = Kb + (size_t)bh * SEQ * HD;
  const __bf16* Vbase = Vt + (size_t)bh * HD * SEQ;

  const v8f vzero = {0.f, 0.f, 0.f, 0.f, 0.f, 0.f, 0.f, 0.f};
  float rm[8], rl[8];
  v8f oc[4];
  for (int i = 0; i < 8; ++i) { rm[i] = -1e30f; rl[i] = 0.f; }
  for (int t = 0; t < 4; ++t) oc[t] = vzero;

  for (int kk0 = 0; kk0 < SEQ; kk0 += 64) {
    if (kk0 + 64 < SEQ) {     // hint next K/V tiles into cache
      __builtin_prefetch(Kbase + (size_t)(kk0 + 64 + ln) * HD, 0, 1);
      __builtin_prefetch(Vbase + (size_t)ln * SEQ + kk0 + 64, 0, 1);
    }

    // ---- scores S[16 x 64] = Qhat * K^T (D=64 in 2 chunks, 4 col tiles) ----
    v8f s[4];
    s[0] = s[1] = s[2] = s[3] = vzero;
    for (int c = 0; c < 2; ++c)
      for (int j = 0; j < 4; ++j) {
        AB kb;   // B frag col n = s index; 32B contiguous per lane
        kb.u8 = *(const v8u*)(Kbase + (size_t)(kk0 + j * 16 + ln) * HD +
                              c * 32 + group * 16);
        s[j] = wmma_bf16(aq[c].bf, kb.bf, s[j]);
      }

    // ---- online softmax (row i lives in one 16-lane half-group) ----
    for (int i = 0; i < 8; ++i) {
      float tm = fmaxf(fmaxf(s[0][i], s[1][i]), fmaxf(s[2][i], s[3][i]));
      tm = hg_max(tm);
      const float nm = fmaxf(rm[i], tm);
      const float sc = EXP2F(rm[i] - nm);
      float e0 = EXP2F(s[0][i] - nm), e1 = EXP2F(s[1][i] - nm);
      float e2 = EXP2F(s[2][i] - nm), e3 = EXP2F(s[3][i] - nm);
      const float ts = hg_sum((e0 + e1) + (e2 + e3));
      rl[i] = rl[i] * sc + ts;
      rm[i] = nm;
      oc[0][i] *= sc; oc[1][i] *= sc; oc[2][i] *= sc; oc[3][i] *= sc;
      // C-layout -> LDS immediately (avoids a 32-register probability array)
      __bf16* pr = &Ptile[wv][group * 8 + i][ln];
      pr[0]  = (__bf16)e0;
      pr[16] = (__bf16)e1;
      pr[32] = (__bf16)e2;
      pr[48] = (__bf16)e3;
    }

    // ---- O[16 x 64] += P * V  (same-wave DS ops are ordered) ----
    for (int c = 0; c < 2; ++c) {
      AB aP;   // A-layout read-back of P chunk c (16 x 32)
      aP.u4[0] = *(const v4u*)&Ptile[wv][ln][c * 32 + group * 8];
      aP.u4[1] = *(const v4u*)&Ptile[wv][ln][c * 32 + 16 + group * 8];
      for (int t = 0; t < 4; ++t) {
        AB bv;
        bv.u4[0] = *(const v4u*)(Vbase + (size_t)(t * 16 + ln) * SEQ + kk0 +
                                 c * 32 + group * 8);
        bv.u4[1] = *(const v4u*)(Vbase + (size_t)(t * 16 + ln) * SEQ + kk0 +
                                 c * 32 + 16 + group * 8);
        oc[t] = wmma_bf16(aP.bf, bv.bf, oc[t]);
      }
    }
  }

  // ---- epilogue: normalize and scatter to [B,S,H*D] fp32 ----
  const int b = bh >> 4, h = bh & 15;
  float* obase = Out + (size_t)(b * SEQ + q0) * HID + h * HD;
  float inv[8];
  for (int i = 0; i < 8; ++i) inv[i] = 1.f / rl[i];
  for (int t = 0; t < 4; ++t)
    for (int i = 0; i < 8; ++i) {
      const int m = group * 8 + i;
      obase[(size_t)m * HID + t * 16 + ln] = oc[t][i] * inv[i];
    }
}

// ---------------------------------------------------------------------------
extern "C" void kernel_launch(void* const* d_in, const int* in_sizes, int n_in,
                              void* d_out, int out_size, void* d_ws,
                              size_t ws_size, hipStream_t stream) {
  const float* X  = (const float*)d_in[0];
  const float* Wq = (const float*)d_in[1];
  const float* bq = (const float*)d_in[2];
  const float* Wk = (const float*)d_in[3];
  const float* bk = (const float*)d_in[4];
  const float* Wv = (const float*)d_in[5];
  const float* bv = (const float*)d_in[6];

  // workspace: Q, K as [BH,S,D] bf16, V transposed [BH,D,S] bf16 (8MB each)
  __bf16* Qb = (__bf16*)d_ws;
  __bf16* Kb = Qb + (size_t)TOK * HID;
  __bf16* Vt = Kb + (size_t)TOK * HID;

  dim3 g1(HID / 64, TOK / 128, 3);         // 16 x 32 x 3 blocks
  qkv_proj_kernel<<<g1, 256, 0, stream>>>(X, Wq, bq, Wk, bk, Wv, bv,
                                          Qb, Kb, Vt);

  dim3 g2((BATCH * NH * (SEQ / 16)) / 8);  // 4096 waves / 8 per block = 512
  attn_kernel<<<g2, 256, 0, stream>>>(Qb, Kb, Vt, (float*)d_out);
}
